// DotProductAttentionDistance_84920093376673
// MI455X (gfx1250) — matrix operand order repaired and checked by reading, not compiled, not run on hardware
//
#include <hip/hip_runtime.h>
#include <hip/hip_bf16.h>
#include <math.h>

typedef __attribute__((ext_vector_type(16))) _Float16 v16h;
typedef __attribute__((ext_vector_type(4)))  _Float16 v4h;
typedef __attribute__((ext_vector_type(8)))  float    v8f;

#define B_   16
#define Q_   2048
#define K_   2048
#define D_   128
#define DV_  128
#define KTILE 64
#define NWAVES 8
#define ROWS (NWAVES * 16)                 // 128 q-rows per block
// 1/sqrt(3*D/2) = 1/sqrt(192), with log2(e) folded in so softmax runs in
// log2 domain and exponentials are raw v_exp_f32.
#define LOG2E      1.4426950408889634f
#define INV_SCALE  0.07216878364870322f
#define SCALE_L2   (INV_SCALE * LOG2E)

union Frag { v16h v; uint4 u[2]; };

// ---------------- prep: Q * (inv_scale*log2e) -> f16 ----------------
__global__ void prep_q(const float* __restrict__ q, _Float16* __restrict__ qh) {
  int i = blockIdx.x * 256 + threadIdx.x;          // 524288 threads, 8 elems each
  const float4* s = (const float4*)q;
  float4 a = s[2 * i], c = s[2 * i + 1];
  union { v4h h[2]; uint4 u; } o;
  o.h[0][0] = (_Float16)(a.x * SCALE_L2);
  o.h[0][1] = (_Float16)(a.y * SCALE_L2);
  o.h[0][2] = (_Float16)(a.z * SCALE_L2);
  o.h[0][3] = (_Float16)(a.w * SCALE_L2);
  o.h[1][0] = (_Float16)(c.x * SCALE_L2);
  o.h[1][1] = (_Float16)(c.y * SCALE_L2);
  o.h[1][2] = (_Float16)(c.z * SCALE_L2);
  o.h[1][3] = (_Float16)(c.w * SCALE_L2);
  ((uint4*)qh)[i] = o.u;
}

// ------- prep: K -> f16, kbias = (D/2 - ||k||^2/2)*inv_scale*log2e -------
__global__ void prep_k(const float* __restrict__ ksrc, _Float16* __restrict__ kh,
                       float* __restrict__ kb) {
  int row  = blockIdx.x * 8 + (threadIdx.x >> 5);  // one wave per key row
  int lane = threadIdx.x & 31;
  float4 x = ((const float4*)ksrc)[row * (D_ / 4) + lane];
  float ss = x.x * x.x + x.y * x.y + x.z * x.z + x.w * x.w;
  union { v4h h; uint2 u; } o;
  o.h[0] = (_Float16)x.x; o.h[1] = (_Float16)x.y;
  o.h[2] = (_Float16)x.z; o.h[3] = (_Float16)x.w;
  *(uint2*)(kh + (size_t)row * D_ + lane * 4) = o.u;
  for (int off = 16; off > 0; off >>= 1) ss += __shfl_xor(ss, off, 32);
  if (lane == 0) kb[row] = (0.5f * (float)D_ - 0.5f * ss) * SCALE_L2;
}

// ---------------- prep: V -> f16 transposed to [B][DV][K] ----------------
__global__ void prep_vt(const float* __restrict__ v, _Float16* __restrict__ vt) {
  __shared__ _Float16 sT[32 * (DV_ + 8)];
  int b  = blockIdx.x >> 6;                        // 64 k-tiles(32) per batch
  int k0 = (blockIdx.x & 63) << 5;
  const float4* src = (const float4*)(v + ((size_t)b * K_ + k0) * DV_);
  for (int j = 0; j < 4; ++j) {
    int idx = threadIdx.x + j * 256;
    int row = idx >> 5, c4 = idx & 31;
    float4 x = src[idx];
    union { v4h h; uint2 u; } o;
    o.h[0] = (_Float16)x.x; o.h[1] = (_Float16)x.y;
    o.h[2] = (_Float16)x.z; o.h[3] = (_Float16)x.w;
    *(uint2*)&sT[row * (DV_ + 8) + c4 * 4] = o.u;
  }
  __syncthreads();
  int dv = threadIdx.x >> 1, half = threadIdx.x & 1;
  union { _Float16 h[16]; uint4 u[2]; } o;
  for (int j = 0; j < 16; ++j) o.h[j] = sT[(half * 16 + j) * (DV_ + 8) + dv];
  uint4* dst = (uint4*)(vt + ((size_t)b * DV_ + dv) * K_ + k0 + half * 16);
  dst[0] = o.u[0];
  dst[1] = o.u[1];
}

// ---------------- main: flash attention, f16 WMMA + async LDS ----------------
__global__ __launch_bounds__(256, 1)
void attn_main(const _Float16* __restrict__ qh, const _Float16* __restrict__ kh,
               const _Float16* __restrict__ vt, const float* __restrict__ kbias,
               const int* __restrict__ vlens, float* __restrict__ out) {
  __shared__ _Float16 sK[2][KTILE * D_];           // 32 KB  (keys row-major)
  __shared__ _Float16 sV[2][DV_ * KTILE];          // 32 KB  (V transposed: [dv][k])
  __shared__ _Float16 sP[NWAVES][16 * KTILE];      // 16 KB  per-wave P transpose
  __shared__ int sKend;

  const int b    = blockIdx.y;
  const int q0   = blockIdx.x * ROWS;
  const int tid  = threadIdx.x;
  const int lane = tid & 31, w = tid >> 5;
  const int n    = lane & 15, h = lane >> 4;

  if (tid == 0) sKend = 1;
  __syncthreads();
  if (tid < ROWS) atomicMax(&sKend, vlens[b * Q_ + q0 + tid]);
  __syncthreads();
  const int ntiles = (sKend + KTILE - 1) / KTILE;

  int vl[8];
  const int rowbase = q0 + w * 16 + 8 * h;
  for (int r = 0; r < 8; ++r) vl[r] = vlens[b * Q_ + rowbase + r];

  // wave-level max valid_len over its 16 rows (full 32-lane reduce: halves differ)
  int wvmax = vl[0];
  for (int r = 1; r < 8; ++r) wvmax = max(wvmax, vl[r]);
  for (int off = 1; off < 32; off <<= 1)
    wvmax = max(wvmax, __shfl_xor(wvmax, off, 32));

  // resident Q A-fragments (16x32 f16), contiguous layout: dims 16h..16h+15
  v16h qf[4];
  {
    const _Float16* qrow = qh + ((size_t)b * Q_ + q0 + w * 16 + n) * D_;
    for (int c = 0; c < 4; ++c) {
      Frag f;
      f.u[0] = *(const uint4*)(qrow + 32 * c + 16 * h);
      f.u[1] = *(const uint4*)(qrow + 32 * c + 16 * h + 8);
      qf[c] = f.v;
    }
  }

  v8f zero = {};
  v8f acc[8];
  for (int t = 0; t < 8; ++t) acc[t] = zero;
  float m[8], l[8];
  for (int r = 0; r < 8; ++r) { m[r] = -1e30f; l[r] = 0.0f; }

  const _Float16* kbase = kh + (size_t)b * K_ * D_;
  const _Float16* vbase = vt + (size_t)b * DV_ * K_;
  const float*    kb    = kbias + b * K_;

  // ---- CDNA5 async copy: global -> LDS, tracked by ASYNCcnt ----
  auto async_tile = [&](int t, int buf) {
    const unsigned k0 = (unsigned)t * KTILE;
#pragma unroll
    for (int j = 0; j < 4; ++j) {                  // K tile: 64 x 128 f16 = 16 KB
      unsigned idx = tid + j * 256;
      unsigned row = idx >> 4, c16 = idx & 15;
      unsigned voff = ((k0 + row) * D_ + c16 * 8) * 2u;
      unsigned lds  = (unsigned)(uintptr_t)&sK[buf][row * D_ + c16 * 8];
      asm volatile("global_load_async_to_lds_b128 %0, %1, %2"
                   :: "v"(lds), "v"(voff), "s"(kbase) : "memory");
    }
#pragma unroll
    for (int j = 0; j < 4; ++j) {                  // Vt tile: 128 x 64 f16 = 16 KB
      unsigned idx = tid + j * 256;
      unsigned dv = idx >> 3, c8 = idx & 7;
      unsigned voff = (dv * K_ + k0 + c8 * 8) * 2u;
      unsigned lds  = (unsigned)(uintptr_t)&sV[buf][dv * KTILE + c8 * 8];
      asm volatile("global_load_async_to_lds_b128 %0, %1, %2"
                   :: "v"(lds), "v"(voff), "s"(vbase) : "memory");
    }
  };
  auto ldfrag = [](const _Float16* p) {
    Frag f;
    f.u[0] = *(const uint4*)p;
    f.u[1] = *(const uint4*)(p + 8);
    return f;
  };

  async_tile(0, 0);
  asm volatile("s_wait_asynccnt 0" ::: "memory");
  __syncthreads();

  float bias[4];                                   // pipelined column biases
  for (int j = 0; j < 4; ++j) bias[j] = kb[16 * j + n];

  for (int t = 0; t < ntiles; ++t) {
    const int cur = t & 1;
    const int k0  = t * KTILE;
    if (t + 1 < ntiles) async_tile(t + 1, cur ^ 1);
    float nb[4] = {0.f, 0.f, 0.f, 0.f};
    if (t + 1 < ntiles)
      for (int j = 0; j < 4; ++j) nb[j] = kb[k0 + KTILE + 16 * j + n];
    if (t + 2 < ntiles) {                          // L2 prefetch two tiles ahead
      __builtin_prefetch(kbase + (size_t)(k0 + 2 * KTILE) * D_ + (size_t)tid * 64, 0, 0);
      if (tid < 128)
        __builtin_prefetch(vbase + (size_t)tid * K_ + k0 + 2 * KTILE, 0, 0);
    }

    if (k0 < wvmax) {                              // wave-uniform: tile has work
      // ---- S = Q K^T : 4 N-subtiles, bias in C, pipelined DS loads ----
      v8f s[4];
      for (int j = 0; j < 4; ++j)
        for (int r = 0; r < 8; ++r) s[j][r] = bias[j];
#pragma unroll
      for (int p2 = 0; p2 < 2; ++p2) {             // subtile pairs (0,1) (2,3)
        const int nA = 32 * p2 + n, nB = 32 * p2 + 16 + n;
        Frag f0 = ldfrag(&sK[cur][nA * D_ + 16 * h]);
        Frag f1 = ldfrag(&sK[cur][nB * D_ + 16 * h]);
#pragma unroll
        for (int c = 0; c < 4; ++c) {
          Frag g0, g1;
          if (c < 3) {
            g0 = ldfrag(&sK[cur][nA * D_ + 32 * (c + 1) + 16 * h]);
            g1 = ldfrag(&sK[cur][nB * D_ + 32 * (c + 1) + 16 * h]);
          }
          s[2 * p2]     = __builtin_amdgcn_wmma_f32_16x16x32_f16(
              false, qf[c], false, f0.v, (short)0, s[2 * p2], false, false);
          s[2 * p2 + 1] = __builtin_amdgcn_wmma_f32_16x16x32_f16(
              false, qf[c], false, f1.v, (short)0, s[2 * p2 + 1], false, false);
          if (c < 3) { f0 = g0; f1 = g1; }
        }
      }

      // ---- mask in place; exp2 of -1e30 underflows to 0, no 2nd select ----
      float mx[8];
      for (int r = 0; r < 8; ++r) {
        for (int j = 0; j < 4; ++j)
          s[j][r] = (k0 + 16 * j + n < vl[r]) ? s[j][r] : -1e30f;
        mx[r] = fmaxf(fmaxf(s[0][r], s[1][r]), fmaxf(s[2][r], s[3][r]));
      }
      for (int off = 1; off < 16; off <<= 1)
        for (int r = 0; r < 8; ++r)
          mx[r] = fmaxf(mx[r], __shfl_xor(mx[r], off, 32));

      // ---- rescale only when the running max actually grew ----
      float dm[8];
      bool grew = false;
      for (int r = 0; r < 8; ++r) {
        float mnew = fmaxf(m[r], mx[r]);
        dm[r] = m[r] - mnew;
        grew |= (mx[r] > m[r]);
        m[r] = mnew;
      }
      if (__any(grew)) {
        for (int r = 0; r < 8; ++r) {
          float alpha = __builtin_amdgcn_exp2f(dm[r]);
          l[r] *= alpha;
          for (int t8 = 0; t8 < 8; ++t8) acc[t8][r] *= alpha;
        }
      }

      float rs[8];
      for (int r = 0; r < 8; ++r) {
        float e0 = __builtin_amdgcn_exp2f(s[0][r] - m[r]);
        float e1 = __builtin_amdgcn_exp2f(s[1][r] - m[r]);
        float e2 = __builtin_amdgcn_exp2f(s[2][r] - m[r]);
        float e3 = __builtin_amdgcn_exp2f(s[3][r] - m[r]);
        rs[r] = (e0 + e1) + (e2 + e3);
        _Float16* pr = &sP[w][(8 * h + r) * KTILE + n];
        pr[0]  = (_Float16)e0;                     // transpose via LDS
        pr[16] = (_Float16)e1;
        pr[32] = (_Float16)e2;
        pr[48] = (_Float16)e3;
      }
      for (int off = 1; off < 16; off <<= 1)
        for (int r = 0; r < 8; ++r) rs[r] += __shfl_xor(rs[r], off, 32);
      for (int r = 0; r < 8; ++r) l[r] += rs[r];

      asm volatile("" ::: "memory");               // keep P stores before loads

      // ---- O += P V : 8 output N-tiles x 2 K-chunks, pipelined DS loads ----
      Frag pa0, pa1;
      pa0.u[0] = *(const uint4*)&sP[w][n * KTILE + 16 * h];
      pa0.u[1] = *(const uint4*)&sP[w][n * KTILE + 8 + 16 * h];
      pa1.u[0] = *(const uint4*)&sP[w][n * KTILE + 32 + 16 * h];
      pa1.u[1] = *(const uint4*)&sP[w][n * KTILE + 40 + 16 * h];
      Frag vf0 = ldfrag(&sV[cur][n * KTILE + 16 * h]);
      Frag vf1 = ldfrag(&sV[cur][n * KTILE + 32 + 16 * h]);
#pragma unroll
      for (int t8 = 0; t8 < 8; ++t8) {
        Frag vg0, vg1;
        if (t8 < 7) {
          vg0 = ldfrag(&sV[cur][(16 * (t8 + 1) + n) * KTILE + 16 * h]);
          vg1 = ldfrag(&sV[cur][(16 * (t8 + 1) + n) * KTILE + 32 + 16 * h]);
        }
        acc[t8] = __builtin_amdgcn_wmma_f32_16x16x32_f16(
            false, pa0.v, false, vf0.v, (short)0, acc[t8], false, false);
        acc[t8] = __builtin_amdgcn_wmma_f32_16x16x32_f16(
            false, pa1.v, false, vf1.v, (short)0, acc[t8], false, false);
        if (t8 < 7) { vf0 = vg0; vf1 = vg1; }
      }
    }

    for (int j = 0; j < 4; ++j) bias[j] = nb[j];
    asm volatile("s_wait_asynccnt 0" ::: "memory");
    __syncthreads();
  }

  // ---- epilogue: O / l ----
  const size_t orow = (size_t)b * Q_ + q0 + w * 16 + 8 * h;
  for (int r = 0; r < 8; ++r) {
    float inv_l = 1.0f / l[r];
    float* op = out + (orow + r) * DV_;
    for (int t8 = 0; t8 < 8; ++t8) op[16 * t8 + n] = acc[t8][r] * inv_l;
  }
}

extern "C" void kernel_launch(void* const* d_in, const int* in_sizes, int n_in,
                              void* d_out, int out_size, void* d_ws, size_t ws_size,
                              hipStream_t stream) {
  const float* q  = (const float*)d_in[0];
  const float* k  = (const float*)d_in[1];
  const float* v  = (const float*)d_in[2];
  const int*   vl = (const int*)d_in[3];
  float* out = (float*)d_out;

  char* ws = (char*)d_ws;                          // 24.125 MB used
  _Float16* qh    = (_Float16*)(ws);
  _Float16* kh    = (_Float16*)(ws + 8ull * 1024 * 1024);
  _Float16* vt    = (_Float16*)(ws + 16ull * 1024 * 1024);
  float*    kbias = (float*)   (ws + 24ull * 1024 * 1024);

  prep_q <<<2048, 256, 0, stream>>>(q, qh);
  prep_k <<<4096, 256, 0, stream>>>(k, kh, kbias);
  prep_vt<<<B_ * (K_ / 32), 256, 0, stream>>>(v, vt);

  dim3 grid(Q_ / ROWS, B_);
  attn_main<<<grid, 256, 0, stream>>>(qh, kh, vt, kbias, vl, out);
}
